// MSAFeaturesAttention_66984309948697
// MI455X (gfx1250) — matrix operand order repaired and checked by reading, not compiled, not run on hardware
//
#include <hip/hip_runtime.h>
#include <math.h>

#define MSA_DIM  256
#define PAIR_DIM 128
#define NHEAD    8
#define HEADDIM  32
#define SEQ      512
#define NB       32
#define NROWS    (NB * SEQ)     // 16384 msa tokens
#define NPAIR    (SEQ * SEQ)    // 262144 pair rows

typedef __bf16 bf16;
typedef __bf16 v16bf __attribute__((ext_vector_type(16)));
typedef __bf16 v8bf  __attribute__((ext_vector_type(8)));
typedef float  v8f   __attribute__((ext_vector_type(8)));

union V16U { v16bf v; v8bf h[2]; };

__device__ __forceinline__ v8f wmma_bf16(v16bf a, v16bf b, v8f c) {
  return __builtin_amdgcn_wmma_f32_16x16x32_bf16(false, a, false, b, (short)0, c,
                                                 false, false);
}

__device__ __forceinline__ float wave_sum32(float x) {
#pragma unroll
  for (int m = 16; m >= 1; m >>= 1) x += __shfl_xor(x, m, 32);
  return x;
}
__device__ __forceinline__ float half_max16(float x) {
#pragma unroll
  for (int m = 8; m >= 1; m >>= 1) x = fmaxf(x, __shfl_xor(x, m, 16));
  return x;
}
__device__ __forceinline__ float half_sum16(float x) {
#pragma unroll
  for (int m = 8; m >= 1; m >>= 1) x += __shfl_xor(x, m, 16);
  return x;
}

// ---------------------------------------------------------------------------
// Weight convert + transpose: Wt[n][k] = bf16(W[k][n]), 256x256.
// ---------------------------------------------------------------------------
__global__ void msa_wconv_kernel(const float* __restrict__ src, bf16* __restrict__ dst) {
  int k = blockIdx.x;
  int n = threadIdx.x;
  dst[(size_t)n * 256 + k] = (bf16)src[(size_t)k * 256 + n];
}

// ---------------------------------------------------------------------------
// MSA layernorm -> bf16.  One wave per 256-wide row, 8 elems/lane.
// ---------------------------------------------------------------------------
__global__ __launch_bounds__(256) void msa_ln_kernel(
    const float* __restrict__ x, const float* __restrict__ g,
    const float* __restrict__ b, bf16* __restrict__ out) {
  int row  = blockIdx.x * 8 + (threadIdx.x >> 5);
  int lane = threadIdx.x & 31;
  const float* xr = x + (size_t)row * MSA_DIM + lane * 8;
  float v[8], s = 0.f, ss = 0.f;
#pragma unroll
  for (int i = 0; i < 8; ++i) { v[i] = xr[i]; s += v[i]; ss += v[i] * v[i]; }
  s  = wave_sum32(s);
  ss = wave_sum32(ss);
  float mu   = s * (1.f / MSA_DIM);
  float var  = ss * (1.f / MSA_DIM) - mu * mu;
  float rstd = rsqrtf(var + 1e-5f);
  bf16* orow = out + (size_t)row * MSA_DIM + lane * 8;
#pragma unroll
  for (int i = 0; i < 8; ++i) {
    int d = lane * 8 + i;
    orow[i] = (bf16)((v[i] - mu) * rstd * g[d] + b[d]);
  }
}

// ---------------------------------------------------------------------------
// Pair bias: layernorm(pair[i,j,:128]) @ wz[128,8] -> bias[h][i][j] (fp32).
// One wave per pair row, 4 elems/lane.  Memory-bound (reads 128 MB).
// ---------------------------------------------------------------------------
__global__ __launch_bounds__(256) void msa_pairbias_kernel(
    const float* __restrict__ pair, const float* __restrict__ g,
    const float* __restrict__ b, const float* __restrict__ wz,
    float* __restrict__ bias) {
  int row  = blockIdx.x * 8 + (threadIdx.x >> 5);
  int lane = threadIdx.x & 31;
  const float* xr = pair + (size_t)row * PAIR_DIM + lane * 4;
  float v[4], s = 0.f, ss = 0.f;
#pragma unroll
  for (int i = 0; i < 4; ++i) { v[i] = xr[i]; s += v[i]; ss += v[i] * v[i]; }
  s  = wave_sum32(s);
  ss = wave_sum32(ss);
  float mu   = s * (1.f / PAIR_DIM);
  float var  = ss * (1.f / PAIR_DIM) - mu * mu;
  float rstd = rsqrtf(var + 1e-5f);
  float p[NHEAD];
#pragma unroll
  for (int h = 0; h < NHEAD; ++h) p[h] = 0.f;
#pragma unroll
  for (int i = 0; i < 4; ++i) {
    int d = lane * 4 + i;
    float t = (v[i] - mu) * rstd * g[d] + b[d];
#pragma unroll
    for (int h = 0; h < NHEAD; ++h) p[h] += t * wz[d * NHEAD + h];
  }
#pragma unroll
  for (int h = 0; h < NHEAD; ++h) p[h] = wave_sum32(p[h]);
  if (lane == 0) {
#pragma unroll
    for (int h = 0; h < NHEAD; ++h) bias[(size_t)h * NPAIR + row] = p[h];
  }
}

// ---------------------------------------------------------------------------
// bf16 WMMA GEMM:  D[16384x256] = A[16384x256] @ Wt^T  (Wt stored [N,K] bf16)
// 8 waves / block, each wave one 16x16 output tile; K-loop of 8 WMMAs.
// mode: 0=q(bf16) 1=k(bf16) 2=v(bf16, transposed [b,h,c,l]) 3=gate(sigmoid,f32)
//       4=out-proj (+bo, f32)
// ---------------------------------------------------------------------------
__global__ __launch_bounds__(256) void msa_gemm_kernel(
    const bf16* __restrict__ A, const bf16* __restrict__ Wt,
    const float* __restrict__ bvec, void* __restrict__ outp, int mode) {
  int wave = threadIdx.x >> 5;
  int lane = threadIdx.x & 31;
  int half = lane >> 4;
  int ln   = lane & 15;
  int row0  = blockIdx.x * 16;
  int ncol0 = blockIdx.y * 128 + wave * 16;

  const bf16* arow = A  + (size_t)(row0 + ln) * MSA_DIM;   // A row (M = ln)
  const bf16* wrow = Wt + (size_t)(ncol0 + ln) * MSA_DIM;  // Wt row (N = ln)

  v8f acc = {0.f, 0.f, 0.f, 0.f, 0.f, 0.f, 0.f, 0.f};
#pragma unroll
  for (int kc = 0; kc < MSA_DIM; kc += 32) {
    V16U a, w;
    // A-frag (16x32 bf16): elems 0..7 = A[m, kc+half*8 ..], 8..15 = A[m, kc+16+half*8 ..]
    a.h[0] = *(const v8bf*)(arow + kc + half * 8);
    a.h[1] = *(const v8bf*)(arow + kc + 16 + half * 8);
    // B-frag (32x16): elems 0..7 = Wt[n][kc+half*8 ..], 8..15 = Wt[n][kc+16+half*8 ..]
    w.h[0] = *(const v8bf*)(wrow + kc + half * 8);
    w.h[1] = *(const v8bf*)(wrow + kc + 16 + half * 8);
    acc = wmma_bf16(a.v, w.v, acc);
  }

#pragma unroll
  for (int r = 0; r < 8; ++r) {
    int row = row0 + r + half * 8;   // D[r + half*8, ln] per ISA C/D layout
    int col = ncol0 + ln;
    float x = acc[r];
    if (mode == 0 || mode == 1) {
      ((bf16*)outp)[(size_t)row * 256 + col] = (bf16)x;
    } else if (mode == 2) {
      int hh = col >> 5, cc = col & 31;
      int bb = row >> 9, ll = row & 511;
      ((bf16*)outp)[(((size_t)(bb * NHEAD + hh) * HEADDIM + cc) << 9) + ll] = (bf16)x;
    } else if (mode == 3) {
      float t = x + bvec[col];
      ((float*)outp)[(size_t)row * 256 + col] = 1.f / (1.f + __expf(-t));
    } else {
      ((float*)outp)[(size_t)row * 256 + col] = x + bvec[col];
    }
  }
}

// ---------------------------------------------------------------------------
// Flash attention: one wave handles a 16-row q-tile for one (b,h).
// S-tile = single v_wmma_f32_16x16x32_bf16 (K = C = 32).  Online softmax.
// P bounced through LDS (f32->bf16) to re-shape C-layout -> A-frag layout.
// ---------------------------------------------------------------------------
__global__ __launch_bounds__(128) void msa_attn_kernel(
    const bf16* __restrict__ q, const bf16* __restrict__ k,
    const bf16* __restrict__ vt, const float* __restrict__ bias,
    const float* __restrict__ gate, bf16* __restrict__ attn) {
  __shared__ __align__(16) bf16 lds_p[4][16][32];
  int wave = threadIdx.x >> 5;
  int lane = threadIdx.x & 31;
  int half = lane >> 4;
  int ln   = lane & 15;
  int b  = blockIdx.z;
  int h  = blockIdx.y;
  int q0 = (blockIdx.x * 4 + wave) * 16;
  const float scale = 0.17677669529663687f; // 1/sqrt(32)

  // Q A-fragment (16x32): lane row = ln, contiguous K = half*16 + e
  const bf16* qrow = q + ((size_t)(b * SEQ + q0 + ln) * NHEAD + h) * HEADDIM;
  V16U aq;
  aq.h[0] = *(const v8bf*)(qrow + half * 16);
  aq.h[1] = *(const v8bf*)(qrow + half * 16 + 8);

  v8f acc0 = {0.f, 0.f, 0.f, 0.f, 0.f, 0.f, 0.f, 0.f};
  v8f acc1 = {0.f, 0.f, 0.f, 0.f, 0.f, 0.f, 0.f, 0.f};
  float mrow[8], lrow[8];
#pragma unroll
  for (int r = 0; r < 8; ++r) { mrow[r] = -INFINITY; lrow[r] = 0.f; }

  for (int kv0 = 0; kv0 < SEQ; kv0 += 32) {
    // K B-fragments: B[c,n] = K[kv0+n, c]; lane n reads contiguous c
    const bf16* kr0 = k + ((size_t)(b * SEQ + kv0 + ln) * NHEAD + h) * HEADDIM + half * 16;
    const bf16* kr1 = kr0 + (size_t)16 * NHEAD * HEADDIM;
    V16U bk0, bk1;
    bk0.h[0] = *(const v8bf*)(kr0);  bk0.h[1] = *(const v8bf*)(kr0 + 8);
    bk1.h[0] = *(const v8bf*)(kr1);  bk1.h[1] = *(const v8bf*)(kr1 + 8);

    v8f z = {0.f, 0.f, 0.f, 0.f, 0.f, 0.f, 0.f, 0.f};
    v8f s0 = wmma_bf16(aq.v, bk0.v, z);
    v8f s1 = wmma_bf16(aq.v, bk1.v, z);

    float sm0[8], sm1[8], mnew[8], corr[8];
#pragma unroll
    for (int r = 0; r < 8; ++r) {
      int qr = q0 + r + half * 8;
      size_t bb = ((size_t)h * SEQ + qr) * SEQ + kv0 + ln;
      sm0[r] = s0[r] * scale + bias[bb];
      sm1[r] = s1[r] * scale + bias[bb + 16];
      float t = half_max16(fmaxf(sm0[r], sm1[r]));
      mnew[r] = fmaxf(mrow[r], t);
      corr[r] = __expf(mrow[r] - mnew[r]);
      mrow[r] = mnew[r];
    }
#pragma unroll
    for (int r = 0; r < 8; ++r) {
      float p0 = __expf(sm0[r] - mnew[r]);
      float p1 = __expf(sm1[r] - mnew[r]);
      int m = r + half * 8;
      lds_p[wave][m][ln]      = (bf16)p0;
      lds_p[wave][m][16 + ln] = (bf16)p1;
      float rs = half_sum16(p0 + p1);
      lrow[r] = lrow[r] * corr[r] + rs;
      acc0[r] = acc0[r] * corr[r];
      acc1[r] = acc1[r] * corr[r];
    }
    // same-wave LDS RAW: writes above feed the A-frag reads below
    asm volatile("s_wait_dscnt 0" ::: "memory");

    V16U ap;
    ap.h[0] = *(const v8bf*)(&lds_p[wave][ln][half * 16]);
    ap.h[1] = *(const v8bf*)(&lds_p[wave][ln][half * 16 + 8]);

    // V B-fragments from transposed V [b,h,c,l]: lane n = c, contiguous kv
    const bf16* vr0 = vt + ((size_t)(b * NHEAD + h) * HEADDIM + ln) * SEQ + kv0 + half * 16;
    const bf16* vr1 = vr0 + (size_t)16 * SEQ;
    V16U bv0, bv1;
    bv0.h[0] = *(const v8bf*)(vr0);  bv0.h[1] = *(const v8bf*)(vr0 + 8);
    bv1.h[0] = *(const v8bf*)(vr1);  bv1.h[1] = *(const v8bf*)(vr1 + 8);

    acc0 = wmma_bf16(ap.v, bv0.v, acc0);
    acc1 = wmma_bf16(ap.v, bv1.v, acc1);
  }

#pragma unroll
  for (int r = 0; r < 8; ++r) {
    int qr = q0 + r + half * 8;
    float inv = 1.f / lrow[r];
    size_t base = (size_t)(b * SEQ + qr) * 256 + h * HEADDIM;
    attn[base + ln]      = (bf16)(acc0[r] * inv * gate[base + ln]);
    attn[base + 16 + ln] = (bf16)(acc1[r] * inv * gate[base + 16 + ln]);
  }
}

// ---------------------------------------------------------------------------
extern "C" void kernel_launch(void* const* d_in, const int* in_sizes, int n_in,
                              void* d_out, int out_size, void* d_ws, size_t ws_size,
                              hipStream_t stream) {
  const float* msa    = (const float*)d_in[0];
  const float* pair   = (const float*)d_in[1];
  const float* n1g    = (const float*)d_in[2];
  const float* n1b    = (const float*)d_in[3];
  const float* wq     = (const float*)d_in[4];
  const float* wk     = (const float*)d_in[5];
  const float* wv     = (const float*)d_in[6];
  const float* nzg    = (const float*)d_in[7];
  const float* nzb    = (const float*)d_in[8];
  const float* wz     = (const float*)d_in[9];
  const float* wg     = (const float*)d_in[10];
  const float* bg     = (const float*)d_in[11];
  const float* wo     = (const float*)d_in[12];
  const float* bo     = (const float*)d_in[13];

  char* ws = (char*)d_ws;
  size_t off = 0;
  auto take = [&](size_t bytes) { char* p = ws + off; off += (bytes + 255) & ~(size_t)255; return p; };

  bf16*  m_bf    = (bf16*)take((size_t)NROWS * MSA_DIM * 2);           // 8 MB
  bf16*  q_bf    = (bf16*)take((size_t)NROWS * MSA_DIM * 2);           // 8 MB
  bf16*  k_bf    = (bf16*)take((size_t)NROWS * MSA_DIM * 2);           // 8 MB
  bf16*  v_t     = (bf16*)take((size_t)NROWS * MSA_DIM * 2);           // 8 MB
  float* gatebuf = (float*)take((size_t)NROWS * MSA_DIM * 4);          // 16 MB
  bf16*  attnbuf = (bf16*)take((size_t)NROWS * MSA_DIM * 2);           // 8 MB
  float* biasbuf = (float*)take((size_t)NHEAD * NPAIR * 4);            // 8 MB
  bf16*  wq_t    = (bf16*)take((size_t)256 * 256 * 2);
  bf16*  wk_t    = (bf16*)take((size_t)256 * 256 * 2);
  bf16*  wv_t    = (bf16*)take((size_t)256 * 256 * 2);
  bf16*  wg_t    = (bf16*)take((size_t)256 * 256 * 2);
  bf16*  wo_t    = (bf16*)take((size_t)256 * 256 * 2);
  (void)ws_size; (void)n_in; (void)in_sizes; (void)out_size;

  // 1) weights -> bf16, transposed [N,K]
  msa_wconv_kernel<<<256, 256, 0, stream>>>(wq, wq_t);
  msa_wconv_kernel<<<256, 256, 0, stream>>>(wk, wk_t);
  msa_wconv_kernel<<<256, 256, 0, stream>>>(wv, wv_t);
  msa_wconv_kernel<<<256, 256, 0, stream>>>(wg, wg_t);
  msa_wconv_kernel<<<256, 256, 0, stream>>>(wo, wo_t);

  // 2) MSA layernorm -> bf16
  msa_ln_kernel<<<NROWS / 8, 256, 0, stream>>>(msa, n1g, n1b, m_bf);

  // 3) pair bias (LN + @wz) -> bias[h][i][j]
  msa_pairbias_kernel<<<NPAIR / 8, 256, 0, stream>>>(pair, nzg, nzb, wz, biasbuf);

  // 4) q / k / v(transposed) / gate projections (bf16 WMMA GEMMs)
  dim3 ggrid(NROWS / 16, 2);
  msa_gemm_kernel<<<ggrid, 256, 0, stream>>>(m_bf, wq_t, nullptr, q_bf, 0);
  msa_gemm_kernel<<<ggrid, 256, 0, stream>>>(m_bf, wk_t, nullptr, k_bf, 1);
  msa_gemm_kernel<<<ggrid, 256, 0, stream>>>(m_bf, wv_t, nullptr, v_t, 2);
  msa_gemm_kernel<<<ggrid, 256, 0, stream>>>(m_bf, wg_t, bg, gatebuf, 3);

  // 5) flash attention with pair bias + gating -> attn bf16 [b,l,h*c]
  dim3 agrid(SEQ / 64, NHEAD, NB);
  msa_attn_kernel<<<agrid, 128, 0, stream>>>(q_bf, k_bf, v_t, biasbuf, gatebuf, attnbuf);

  // 6) output projection (+bo) -> d_out fp32
  msa_gemm_kernel<<<ggrid, 256, 0, stream>>>(attnbuf, wo_t, bo, d_out, 4);
}
